// DynSMHALayer_6236292513898
// MI455X (gfx1250) — matrix operand order repaired and checked by reading, not compiled, not run on hardware
//
#include <hip/hip_runtime.h>
#include <hip/hip_bf16.h>

// ---------------------------------------------------------------------------
// Problem constants (match reference)
// ---------------------------------------------------------------------------
#define B_  4
#define T_  2048
#define C_  2048
#define H_  128
#define E_  16
#define N_  (B_ * T_)          // 8192 tokens

typedef __attribute__((ext_vector_type(16))) __bf16 v16bf;
typedef __attribute__((ext_vector_type(8)))  __bf16 bf16x8;
typedef __attribute__((ext_vector_type(8)))  float  v8f;
typedef __attribute__((ext_vector_type(4)))  unsigned int uint32x4;
typedef __attribute__((ext_vector_type(8)))  int  int32x8;
typedef __attribute__((ext_vector_type(4)))  int  int32x4;

// ---------------------------------------------------------------------------
// WMMA helpers (CDNA5 bf16, D=f32 16x16, K=32)
// ---------------------------------------------------------------------------
__device__ __forceinline__ v8f wmma_bf16(v16bf a, v16bf b, v8f c) {
    return __builtin_amdgcn_wmma_f32_16x16x32_bf16(
        /*neg_a=*/false, a, /*neg_b=*/false, b,
        /*c_mod=*/(short)0, c, /*reuse_a=*/false, /*reuse_b=*/false);
}

// A fragment: 16x32 bf16, row-major source with leading dim ld.
// Layout: lanes 0-15 (row M=lane): K = 0..15;
//         lanes 16-31 (row M=lane-16): K = 16..31.
__device__ __forceinline__ v16bf load_a_frag(const __bf16* src, int ld,
                                             int row0, int k0, int lane) {
    int r = lane & 15, half = lane >> 4;
    const __bf16* p = src + (size_t)(row0 + r) * ld + k0 + half * 16;
    bf16x8 c0 = *(const bf16x8*)(p);
    bf16x8 c1 = *(const bf16x8*)(p + 8);
    v16bf a;
#pragma unroll
    for (int i = 0; i < 8; ++i) { a[i] = c0[i]; a[8 + i] = c1[i]; }
    return a;
}

// B fragment: 32x16 bf16. Source stored "N-major": row n holds K contiguous,
// leading dim ld. Layout used here (same k order as A): lane n (0-15): K=0..15 packed 2-per-VGPR,
// lanes 16-31: K=16..31.
__device__ __forceinline__ v16bf load_b_frag(const __bf16* src, int ld,
                                             int n0, int k0, int lane) {
    int n = lane & 15, half = lane >> 4;
    const __bf16* p = src + (size_t)(n0 + n) * ld + k0 + half * 16;
    bf16x8 c0 = *(const bf16x8*)(p);
    bf16x8 c1 = *(const bf16x8*)(p + 8);
    v16bf b;
#pragma unroll
    for (int i = 0; i < 8; ++i) { b[i] = c0[i]; b[8 + i] = c1[i]; }
    return b;
}

// B fragment from an LDS-resident 128x32 tile (row n: 32 K-contiguous bf16).
__device__ __forceinline__ v16bf load_b_frag_lds(const __bf16* tile,
                                                 int n0, int lane) {
    int n = lane & 15, half = lane >> 4;
    const __bf16* p = tile + (n0 + n) * 32 + half * 16;
    bf16x8 c0 = *(const bf16x8*)(p);
    bf16x8 c1 = *(const bf16x8*)(p + 8);
    v16bf b;
#pragma unroll
    for (int i = 0; i < 8; ++i) { b[i] = c0[i]; b[8 + i] = c1[i]; }
    return b;
}

// ---------------------------------------------------------------------------
// Tensor Data Mover: async 2-D tile load (rows x 32 bf16) global -> LDS.
// D# packing per CDNA5 ISA ch.8 (group0 128b, group1 256b; groups 2/3 zero).
// ---------------------------------------------------------------------------
__device__ __forceinline__ void tdm_load_tile_2d(unsigned lds_off,
                                                 const __bf16* gsrc,
                                                 unsigned tensor_d0,   // elems
                                                 unsigned tensor_d1,   // rows
                                                 unsigned tile_d0,     // elems
                                                 unsigned tile_d1,     // rows
                                                 unsigned stride0) {   // elems
    unsigned long long ga = (unsigned long long)gsrc;
    uint32x4 g0;
    g0[0] = 1u;                                        // count=1 (user D#)
    g0[1] = lds_off;                                   // lds_addr (bytes)
    g0[2] = (unsigned)(ga & 0xFFFFFFFFu);              // global_addr[31:0]
    g0[3] = (unsigned)((ga >> 32) & 0x01FFFFFFu)       // global_addr[56:32]
          | (2u << 30);                                // type=2 ("image")
    int32x8 g1;
    g1[0] = (int)(1u << 16);                           // data_size=1 -> 2 bytes
    g1[1] = (int)(tensor_d0 << 16);                    // tensor_dim0[15:0]
    g1[2] = (int)((tensor_d0 >> 16) | (tensor_d1 << 16));
    g1[3] = (int)((tensor_d1 >> 16) | (tile_d0 << 16));// tile_dim0
    g1[4] = (int)tile_d1;                              // tile_dim1 (tile_dim2=0)
    g1[5] = (int)stride0;                              // dim0_stride[31:0]
    g1[6] = 0;                                         // dim0_stride[47:32]
    g1[7] = 0;
    int32x4 z4 = {0, 0, 0, 0};
#if defined(__clang_major__) && (__clang_major__ >= 23)
    int32x8 z8 = {0, 0, 0, 0, 0, 0, 0, 0};
    __builtin_amdgcn_tensor_load_to_lds(g0, g1, z4, z4, z8, 0);
#else
    __builtin_amdgcn_tensor_load_to_lds(g0, g1, z4, z4, 0);
#endif
}

// ---------------------------------------------------------------------------
// Kernel: per-expert column norms of sim_matrix -> 1/max(norm, 1e-12)
// ---------------------------------------------------------------------------
__global__ void colnorm_kernel(const float* __restrict__ sim,
                               float* __restrict__ inv_coln) {
    __shared__ float red[256];
    int e = blockIdx.x;
    float s = 0.f;
    for (int c = threadIdx.x; c < C_; c += 256) {
        float v = sim[(size_t)c * E_ + e];
        s += v * v;
    }
    red[threadIdx.x] = s;
    __syncthreads();
    for (int off = 128; off > 0; off >>= 1) {
        if (threadIdx.x < off) red[threadIdx.x] += red[threadIdx.x + off];
        __syncthreads();
    }
    if (threadIdx.x == 0)
        inv_coln[e] = 1.f / fmaxf(sqrtf(red[0]), 1e-12f);
}

// ---------------------------------------------------------------------------
// Kernel: gating. One wave (32 lanes) per token.
// ---------------------------------------------------------------------------
__global__ __launch_bounds__(32)
void gating_kernel(const float* __restrict__ x, const float* __restrict__ sim,
                   const float* __restrict__ gates,
                   const float* __restrict__ inv_coln,
                   float* __restrict__ we) {
    int n = blockIdx.x;
    int lane = threadIdx.x;
    __shared__ float part[E_ * 32];

    float dot[E_];
#pragma unroll
    for (int e = 0; e < E_; ++e) dot[e] = 0.f;
    float nrm2 = 0.f;

    const float* xr = x + (size_t)n * C_;
    for (int c = lane; c < C_; c += 32) {
        float xv = xr[c];
        nrm2 += xv * xv;
        const float4* srow = (const float4*)(sim + (size_t)c * E_);
#pragma unroll
        for (int q4 = 0; q4 < 4; ++q4) {
            float4 s = srow[q4];
            dot[q4 * 4 + 0] += xv * s.x;
            dot[q4 * 4 + 1] += xv * s.y;
            dot[q4 * 4 + 2] += xv * s.z;
            dot[q4 * 4 + 3] += xv * s.w;
        }
    }
#pragma unroll
    for (int e = 0; e < E_; ++e) part[e * 32 + lane] = dot[e];
#pragma unroll
    for (int m = 16; m > 0; m >>= 1) nrm2 += __shfl_xor(nrm2, m);
    __syncthreads();

    if (lane < E_) {
        float d = 0.f;
#pragma unroll 8
        for (int i = 0; i < 32; ++i) d += part[lane * 32 + i];
        float invx  = 1.f / fmaxf(sqrtf(nrm2), 1e-12f);
        float g     = gates[lane];
        float sig   = 1.f / (1.f + __expf(-g));
        float logit = d * invx * inv_coln[lane] - sig;
        float gated = fmaxf(logit, 0.f);
        bool  hard  = (logit > 0.f);

        int nActive = (int)__popcll(__ballot(hard));

        float v1 = logit; int i1 = lane;
#pragma unroll
        for (int m = 1; m <= 8; m <<= 1) {
            float ov = __shfl_xor(v1, m);
            int   oi = __shfl_xor(i1, m);
            if (ov > v1 || (ov == v1 && oi < i1)) { v1 = ov; i1 = oi; }
        }
        float v2 = (lane == i1) ? -3e38f : logit;
        int   i2 = (lane == i1) ? 1000   : lane;
#pragma unroll
        for (int m = 1; m <= 8; m <<= 1) {
            float ov = __shfl_xor(v2, m);
            int   oi = __shfl_xor(i2, m);
            if (ov > v2 || (ov == v2 && oi < i2)) { v2 = ov; i2 = oi; }
        }

        bool msk = (nActive == 0) ? (lane == i1 || lane == i2) : hard;
        float ml = msk ? gated : -1e30f;
        float mx = ml;
#pragma unroll
        for (int m = 1; m <= 8; m <<= 1) mx = fmaxf(mx, __shfl_xor(mx, m));
        float p = msk ? __expf(ml - mx) : 0.f;
        float s = p;
#pragma unroll
        for (int m = 1; m <= 8; m <<= 1) s += __shfl_xor(s, m);
        we[(size_t)n * E_ + lane] = p / s;
    }
}

// ---------------------------------------------------------------------------
// Kernel: f32 -> bf16 elementwise convert
// ---------------------------------------------------------------------------
__global__ void cvt_bf16_kernel(const float* __restrict__ src,
                                __bf16* __restrict__ dst, size_t total) {
    size_t i = (size_t)blockIdx.x * blockDim.x + threadIdx.x;
    if (i < total) dst[i] = (__bf16)src[i];
}

// ---------------------------------------------------------------------------
// Kernel: transpose last two dims + convert to bf16.
// src [E, R, Cc] (f32) -> dst [E, Cc, R] (bf16)
// ---------------------------------------------------------------------------
__global__ void transpose_w_kernel(const float* __restrict__ src,
                                   __bf16* __restrict__ dst, int R, int Cc) {
    size_t idx = (size_t)blockIdx.x * blockDim.x + threadIdx.x;
    size_t total = (size_t)E_ * R * Cc;
    if (idx >= total) return;
    int cc = (int)(idx % Cc);
    size_t t = idx / Cc;
    int r = (int)(t % R);
    int e = (int)(t / R);
    dst[((size_t)e * Cc + cc) * R + r] = (__bf16)src[idx];
}

// ---------------------------------------------------------------------------
// Kernel: grouped expert GEMM for Q/K/V with TDM-staged weights.
//   out[n,h] = sum_e we[n,e] * (x[n,:] @ W_e[:,h]);  W stored transposed [E,H,C]
// Block = 128 threads (4 waves); waves cover 64 rows x 128 cols.
// Weight tile (128 h-rows x 32 K) streamed into LDS by the Tensor Data Mover,
// double-buffered; all 4 waves consume it via ds_load B-fragments.
// ---------------------------------------------------------------------------
__global__ __launch_bounds__(128)
void qkv_gemm_kernel(const __bf16* __restrict__ x_bf,
                     const __bf16* __restrict__ qwT,
                     const __bf16* __restrict__ kwT,
                     const __bf16* __restrict__ vwT,
                     const float* __restrict__ we,
                     float* __restrict__ qf, float* __restrict__ kf,
                     float* __restrict__ vf) {
    const __bf16* wT;
    float* out;
    if (blockIdx.y == 0)      { wT = qwT; out = qf; }
    else if (blockIdx.y == 1) { wT = kwT; out = kf; }
    else                      { wT = vwT; out = vf; }

    __shared__ __bf16 bbuf[2][H_ * 32];     // 2 x 8 KB weight tiles

    int wave = threadIdx.x >> 5, lane = threadIdx.x & 31;
    int row0 = blockIdx.x * 64 + wave * 16;
    int half = lane >> 4, col = lane & 15;

    const int KCH = C_ / 32;                // 64 k-chunks per expert
    const int TOTAL = E_ * KCH;             // 1024 chunks streamed per block

    // Prologue: DMA chunk 0 into buffer 0 (wave 0 only issues TDM).
    if (threadIdx.x < 32) {
        tdm_load_tile_2d((unsigned)(unsigned long long)&bbuf[0][0],
                         wT, C_, H_, 32, H_, C_);
    }

    v8f fin[8];
#pragma unroll
    for (int ct = 0; ct < 8; ++ct) fin[ct] = (v8f)(0.f);

    int chunk = 0;
    for (int e = 0; e < E_; ++e) {
        v8f acc[8];
#pragma unroll
        for (int ct = 0; ct < 8; ++ct) acc[ct] = (v8f)(0.f);

        for (int kci = 0; kci < KCH; ++kci) {
            int kc = kci * 32;
            int nxt = chunk + 1;
            if (threadIdx.x < 32) {
                if (nxt < TOTAL) {
                    int e2  = nxt >> 6;          // KCH == 64
                    int kc2 = (nxt & 63) * 32;
                    tdm_load_tile_2d(
                        (unsigned)(unsigned long long)&bbuf[nxt & 1][0],
                        wT + (size_t)e2 * H_ * C_ + kc2, C_, H_, 32, H_, C_);
                    __builtin_amdgcn_s_wait_tensorcnt(1); // oldest (cur) done
                } else {
                    __builtin_amdgcn_s_wait_tensorcnt(0);
                }
            }
            __syncthreads();                     // tile[chunk] visible to all

            const __bf16* tile = &bbuf[chunk & 1][0];
            v16bf a = load_a_frag(x_bf, C_, row0, kc, lane);
#pragma unroll
            for (int ct = 0; ct < 8; ++ct) {
                v16bf b = load_b_frag_lds(tile, ct * 16, lane);
                acc[ct] = wmma_bf16(a, b, acc[ct]);
            }
            __syncthreads();                     // done reading before reuse
            chunk = nxt;
        }
#pragma unroll
        for (int i = 0; i < 8; ++i) {
            float wv = we[(size_t)(row0 + i + half * 8) * E_ + e];
#pragma unroll
            for (int ct = 0; ct < 8; ++ct) fin[ct][i] += wv * acc[ct][i];
        }
    }
#pragma unroll
    for (int ct = 0; ct < 8; ++ct)
#pragma unroll
        for (int i = 0; i < 8; ++i)
            out[(size_t)(row0 + i + half * 8) * H_ + ct * 16 + col] = fin[ct][i];
}

// ---------------------------------------------------------------------------
// Kernel: RoPE on q,k; fold 1/sqrt(H) into q; emit bf16.
// ---------------------------------------------------------------------------
__global__ void rope_kernel(const float* __restrict__ qf,
                            const float* __restrict__ kf,
                            __bf16* __restrict__ q_bf,
                            __bf16* __restrict__ k_bf) {
    int idx = blockIdx.x * blockDim.x + threadIdx.x;
    if (idx >= N_ * (H_ / 2)) return;
    int h = idx & (H_ / 2 - 1);
    int n = idx >> 6;                 // H/2 == 64
    int t = n & (T_ - 1);
    const float LOG_BASE = 9.210340371976184f;   // ln(10000)
    float invf = __expf(-((float)(2 * h) / (float)H_) * LOG_BASE);
    float fr = (float)t * invf;
    float cs = __cosf(fr), sn = __sinf(fr);
    const float scale = 0.08838834764831845f;    // 1/sqrt(128)

    size_t base = (size_t)n * H_ + h;
    float q0 = qf[base], q1 = qf[base + H_ / 2];
    q_bf[base]          = (__bf16)((q0 * cs - q1 * sn) * scale);
    q_bf[base + H_ / 2] = (__bf16)((q1 * cs + q0 * sn) * scale);
    float k0 = kf[base], k1 = kf[base + H_ / 2];
    k_bf[base]          = (__bf16)(k0 * cs - k1 * sn);
    k_bf[base + H_ / 2] = (__bf16)(k1 * cs + k0 * sn);
}

// ---------------------------------------------------------------------------
// Kernel: V -> bf16 transposed per batch: vT[b][h][t] = v[b*T+t][h]
// ---------------------------------------------------------------------------
__global__ void vT_kernel(const float* __restrict__ vf,
                          __bf16* __restrict__ vT) {
    size_t idx = (size_t)blockIdx.x * blockDim.x + threadIdx.x;
    if (idx >= (size_t)N_ * H_) return;
    int h = (int)(idx % H_);
    int n = (int)(idx / H_);
    int b = n / T_, t = n % T_;
    vT[((size_t)b * H_ + h) * T_ + t] = (__bf16)vf[idx];
}

// ---------------------------------------------------------------------------
// Kernel: causal flash attention. One wave per 16-row Q tile, 32 keys/step.
// ---------------------------------------------------------------------------
__global__ __launch_bounds__(32)
void attn_kernel(const __bf16* __restrict__ q_bf,
                 const __bf16* __restrict__ k_bf,
                 const __bf16* __restrict__ vT,
                 __bf16* __restrict__ attn_bf) {
    int b = blockIdx.y;
    int q0 = blockIdx.x * 16;
    int lane = threadIdx.x;
    int half = lane >> 4, nIdx = lane & 15;

    __shared__ __bf16 pT[16 * 32];

    const __bf16* qb = q_bf + (size_t)b * T_ * H_;
    const __bf16* kb = k_bf + (size_t)b * T_ * H_;
    const __bf16* vb = vT + (size_t)b * H_ * T_;

    v16bf qa[4];
#pragma unroll
    for (int hc = 0; hc < 4; ++hc)
        qa[hc] = load_a_frag(qb, H_, q0, hc * 32, lane);

    v8f o[8];
#pragma unroll
    for (int ct = 0; ct < 8; ++ct) o[ct] = (v8f)(0.f);
    float m[8], l[8];
#pragma unroll
    for (int i = 0; i < 8; ++i) { m[i] = -1e30f; l[i] = 0.f; }

    int nkb = (q0 + 16 + 31) / 32;
    for (int kbk = 0; kbk < nkb; ++kbk) {
        int kk = kbk * 32;
        v8f s0 = (v8f)(0.f), s1 = (v8f)(0.f);
#pragma unroll
        for (int hc = 0; hc < 4; ++hc) {
            v16bf b0 = load_b_frag(kb, H_, kk,      hc * 32, lane);
            v16bf b1 = load_b_frag(kb, H_, kk + 16, hc * 32, lane);
            s0 = wmma_bf16(qa[hc], b0, s0);
            s1 = wmma_bf16(qa[hc], b1, s1);
        }
        int key0 = kk + nIdx, key1 = kk + 16 + nIdx;
        float al[8];
#pragma unroll
        for (int i = 0; i < 8; ++i) {
            int row = q0 + i + half * 8;
            if (key0 > row) s0[i] = -1e30f;
            if (key1 > row) s1[i] = -1e30f;
            float mv = fmaxf(s0[i], s1[i]);
#pragma unroll
            for (int mm = 1; mm <= 8; mm <<= 1)
                mv = fmaxf(mv, __shfl_xor(mv, mm));
            float mn = fmaxf(m[i], mv);
            al[i] = __expf(m[i] - mn);
            m[i] = mn;
        }
        __syncthreads();
#pragma unroll
        for (int i = 0; i < 8; ++i) {
            float p0 = __expf(s0[i] - m[i]);
            float p1 = __expf(s1[i] - m[i]);
            float rs = p0 + p1;
#pragma unroll
            for (int mm = 1; mm <= 8; mm <<= 1) rs += __shfl_xor(rs, mm);
            l[i] = l[i] * al[i] + rs;
            int row = i + half * 8;
            pT[row * 32 + nIdx]      = (__bf16)p0;
            pT[row * 32 + 16 + nIdx] = (__bf16)p1;
        }
#pragma unroll
        for (int ct = 0; ct < 8; ++ct)
#pragma unroll
            for (int i = 0; i < 8; ++i) o[ct][i] *= al[i];
        __syncthreads();

        v16bf pa;
#pragma unroll
        for (int j = 0; j < 8; ++j) {
            pa[j]     = pT[nIdx * 32 + half * 16 + j];
            pa[8 + j] = pT[nIdx * 32 + half * 16 + 8 + j];
        }
#pragma unroll
        for (int ct = 0; ct < 8; ++ct) {
            v16bf bv = load_b_frag(vb, T_, ct * 16, kk, lane);
            o[ct] = wmma_bf16(pa, bv, o[ct]);
        }
    }
#pragma unroll
    for (int ct = 0; ct < 8; ++ct)
#pragma unroll
        for (int i = 0; i < 8; ++i) {
            size_t row = (size_t)(b * T_ + q0 + i + half * 8);
            attn_bf[row * H_ + ct * 16 + nIdx] = (__bf16)(o[ct][i] / l[i]);
        }
}

// ---------------------------------------------------------------------------
// Kernel: grouped expert output projection with TDM-staged weights.
//   out[n,c] = sum_e we[n,e] * (attn[n,:] @ O_e[:,c]);  O stored [E,C,H]
// Per block: 64 rows x 128 cols; weight tile = 128 c-rows x 32 K (of H).
// ---------------------------------------------------------------------------
__global__ __launch_bounds__(128)
void oproj_gemm_kernel(const __bf16* __restrict__ attn_bf,
                       const __bf16* __restrict__ owT,
                       const float* __restrict__ we,
                       float* __restrict__ out) {
    __shared__ __bf16 bbuf[2][128 * 32];

    int wave = threadIdx.x >> 5, lane = threadIdx.x & 31;
    int row0 = blockIdx.x * 64 + wave * 16;
    int col0 = blockIdx.y * 128;
    int half = lane >> 4, col = lane & 15;

    const int KCH = H_ / 32;                 // 4 k-chunks per expert
    const int TOTAL = E_ * KCH;              // 64 chunks per block

    if (threadIdx.x < 32) {
        tdm_load_tile_2d((unsigned)(unsigned long long)&bbuf[0][0],
                         owT + (size_t)col0 * H_, H_, 128, 32, 128, H_);
    }

    v8f fin[8];
#pragma unroll
    for (int ct = 0; ct < 8; ++ct) fin[ct] = (v8f)(0.f);

    int chunk = 0;
    for (int e = 0; e < E_; ++e) {
        v8f acc[8];
#pragma unroll
        for (int ct = 0; ct < 8; ++ct) acc[ct] = (v8f)(0.f);

        for (int kci = 0; kci < KCH; ++kci) {
            int kc = kci * 32;
            int nxt = chunk + 1;
            if (threadIdx.x < 32) {
                if (nxt < TOTAL) {
                    int e2  = nxt >> 2;       // KCH == 4
                    int kc2 = (nxt & 3) * 32;
                    tdm_load_tile_2d(
                        (unsigned)(unsigned long long)&bbuf[nxt & 1][0],
                        owT + ((size_t)e2 * C_ + col0) * H_ + kc2,
                        H_, 128, 32, 128, H_);
                    __builtin_amdgcn_s_wait_tensorcnt(1);
                } else {
                    __builtin_amdgcn_s_wait_tensorcnt(0);
                }
            }
            __syncthreads();

            const __bf16* tile = &bbuf[chunk & 1][0];
            v16bf a = load_a_frag(attn_bf, H_, row0, kc, lane);
#pragma unroll
            for (int ct = 0; ct < 8; ++ct) {
                v16bf b = load_b_frag_lds(tile, ct * 16, lane);
                acc[ct] = wmma_bf16(a, b, acc[ct]);
            }
            __syncthreads();
            chunk = nxt;
        }
#pragma unroll
        for (int i = 0; i < 8; ++i) {
            float wv = we[(size_t)(row0 + i + half * 8) * E_ + e];
#pragma unroll
            for (int ct = 0; ct < 8; ++ct) fin[ct][i] += wv * acc[ct][i];
        }
    }
#pragma unroll
    for (int ct = 0; ct < 8; ++ct)
#pragma unroll
        for (int i = 0; i < 8; ++i)
            out[(size_t)(row0 + i + half * 8) * C_ + col0 + ct * 16 + col] =
                fin[ct][i];
}

// ---------------------------------------------------------------------------
// Host launcher
// ---------------------------------------------------------------------------
extern "C" void kernel_launch(void* const* d_in, const int* in_sizes, int n_in,
                              void* d_out, int out_size, void* d_ws,
                              size_t ws_size, hipStream_t stream) {
    (void)in_sizes; (void)n_in; (void)out_size; (void)ws_size;

    const float* x      = (const float*)d_in[0];   // [B,T,C]
    /* position_ids d_in[1] == arange(T): folded into RoPE kernel */
    const float* sim    = (const float*)d_in[2];   // [C,E]
    const float* gates  = (const float*)d_in[3];   // [E]
    const float* q_proj = (const float*)d_in[4];   // [E,C,H]
    const float* k_proj = (const float*)d_in[5];   // [E,C,H]
    const float* v_proj = (const float*)d_in[6];   // [E,C,H]
    const float* o_proj = (const float*)d_in[7];   // [E,H,C]
    float* out = (float*)d_out;                    // [B,T,C]

    // ---- workspace carve-up (all offsets 256B aligned) ----
    char* w = (char*)d_ws;
    float* we        = (float*)w;  w += (size_t)N_ * E_ * 4;
    float* inv_coln  = (float*)w;  w += 256;
    __bf16* x_bf     = (__bf16*)w; w += (size_t)N_ * C_ * 2;
    __bf16* qwT      = (__bf16*)w; w += (size_t)E_ * C_ * H_ * 2;
    __bf16* kwT      = (__bf16*)w; w += (size_t)E_ * C_ * H_ * 2;
    __bf16* vwT      = (__bf16*)w; w += (size_t)E_ * C_ * H_ * 2;
    __bf16* owT      = (__bf16*)w; w += (size_t)E_ * C_ * H_ * 2;
    float* qf        = (float*)w;  w += (size_t)N_ * H_ * 4;
    float* kf        = (float*)w;  w += (size_t)N_ * H_ * 4;
    float* vf        = (float*)w;  w += (size_t)N_ * H_ * 4;
    __bf16* q_bf     = (__bf16*)w; w += (size_t)N_ * H_ * 2;
    __bf16* k_bf     = (__bf16*)w; w += (size_t)N_ * H_ * 2;
    __bf16* vT       = (__bf16*)w; w += (size_t)B_ * H_ * T_ * 2;
    __bf16* attn_bf  = (__bf16*)w; w += (size_t)N_ * H_ * 2;

    // 1. gating
    colnorm_kernel<<<E_, 256, 0, stream>>>(sim, inv_coln);
    gating_kernel<<<N_, 32, 0, stream>>>(x, sim, gates, inv_coln, we);

    // 2. prep: x -> bf16; weight transposes (+bf16)
    {
        size_t tot = (size_t)N_ * C_;
        cvt_bf16_kernel<<<(unsigned)((tot + 255) / 256), 256, 0, stream>>>(x, x_bf, tot);
        unsigned tb = (unsigned)(((size_t)E_ * C_ * H_ + 255) / 256);
        transpose_w_kernel<<<tb, 256, 0, stream>>>(q_proj, qwT, C_, H_); // [E,C,H]->[E,H,C]
        transpose_w_kernel<<<tb, 256, 0, stream>>>(k_proj, kwT, C_, H_);
        transpose_w_kernel<<<tb, 256, 0, stream>>>(v_proj, vwT, C_, H_);
        transpose_w_kernel<<<tb, 256, 0, stream>>>(o_proj, owT, H_, C_); // [E,H,C]->[E,C,H]
    }

    // 3. grouped QKV GEMM (WMMA + TDM-staged weights)
    qkv_gemm_kernel<<<dim3(N_ / 64, 3), 128, 0, stream>>>(
        x_bf, qwT, kwT, vwT, we, qf, kf, vf);

    // 4. RoPE + conversions
    rope_kernel<<<(N_ * (H_ / 2) + 255) / 256, 256, 0, stream>>>(qf, kf, q_bf, k_bf);
    vT_kernel<<<(unsigned)(((size_t)N_ * H_ + 255) / 256), 256, 0, stream>>>(vf, vT);

    // 5. causal flash attention (WMMA)
    attn_kernel<<<dim3(T_ / 16, B_), 32, 0, stream>>>(q_bf, k_bf, vT, attn_bf);

    // 6. grouped output projection (WMMA + TDM-staged weights) -> d_out
    oproj_gemm_kernel<<<dim3(N_ / 64, C_ / 128), 128, 0, stream>>>(
        attn_bf, owT, we, out);
}
